// TemporalGCN_7885559955673
// MI455X (gfx1250) — matrix operand order repaired and checked by reading, not compiled, not run on hardware
//
#include <hip/hip_runtime.h>

// ---------------------------------------------------------------------------
// Types for CDNA5 WMMA (wave32): v_wmma_f32_16x16x32_bf16
// ---------------------------------------------------------------------------
typedef __attribute__((ext_vector_type(16))) __bf16 v16bf;
typedef __attribute__((ext_vector_type(8)))  __bf16 v8bf;
typedef __attribute__((ext_vector_type(8)))  float  v8f;

#define HID 128
#define TS  128
#define MB  64                          // rows per block (4 waves of 16)
#define WHH_ELEMS (32 * 4 * 32 * 16)    // 65536 packed bf16 elements (128 KB)

static __device__ __forceinline__ __bf16 f2bf(float f) {
  union { float f; unsigned u; } v; v.f = f;
  unsigned r = v.u + 0x7FFFu + ((v.u >> 16) & 1u);   // round-to-nearest-even
  union { unsigned short s; __bf16 b; } o; o.s = (unsigned short)(r >> 16);
  return o.b;
}
static __device__ __forceinline__ float bf2f(__bf16 b) {
  union { unsigned short s; __bf16 b; } i; i.b = b;
  union { unsigned u; float f; } o; o.u = ((unsigned)i.s) << 16;
  return o.f;
}

// CDNA5 hardware transcendental tanh (ISA lists TANH among TRANS32 ops).
// The embedded s_delay_alu enforces the TRANS->consumer dependency the
// compiler cannot see through inline asm.
static __device__ __forceinline__ float tanh_hw(float x) {
  float r;
  asm("v_tanh_f32 %0, %1\n\ts_delay_alu instid0(TRANS32_DEP_1)"
      : "=v"(r) : "v"(x));
  return r;
}
// sigmoid(x) = 0.5*tanh(0.5x) + 0.5  (mul + tanh + fma)
static __device__ __forceinline__ float sig_hw(float x) {
  return __builtin_fmaf(0.5f, tanh_hw(0.5f * x), 0.5f);
}

// ---------------------------------------------------------------------------
// Pack W_hh [512,128] f32 -> bf16 B-fragments for g = h @ W_hh^T.
// dst[((ntile*4+kc)*32+lane)*16 + j]; lane: n = ntile*16+(lane&15);
// element j holds K = kc*32 + 16*(lane>>4) + j.
// ---------------------------------------------------------------------------
__global__ void pack_whh_kernel(const float* __restrict__ W, __bf16* __restrict__ dst) {
  int idx = blockIdx.x * 256 + threadIdx.x;           // 65536 elements
  int j    = idx & 15;
  int lane = (idx >> 4) & 31;
  int kc   = (idx >> 9) & 3;
  int nt   = idx >> 11;                                // 0..31
  int n = nt * 16 + (lane & 15);                       // gate column 0..511
  int k = kc * 32 + 16 * (lane >> 4) + j;              // 0..127
  dst[idx] = f2bf(W[n * HID + k]);
}

// Pack W [128,128] f32 -> bf16 B-fragments for x @ W.  B[k][n] = W[k][n].
__global__ void pack_w_kernel(const float* __restrict__ W, __bf16* __restrict__ dst) {
  int idx = blockIdx.x * 256 + threadIdx.x;           // 16384 elements
  int j    = idx & 15;
  int lane = (idx >> 4) & 31;
  int kc   = (idx >> 9) & 3;
  int nt   = idx >> 11;                                // 0..7
  int n = nt * 16 + (lane & 15);                       // 0..127
  int k = kc * 32 + 16 * (lane >> 4) + j;              // 0..127
  dst[idx] = f2bf(W[k * HID + n]);
}

// ---------------------------------------------------------------------------
// LSTM over T=128 steps. Block = 128 threads (4 waves); each wave owns a
// private 16-node tile. Packed W_hh staged once into LDS (128 KB); input
// weights/biases in a 4 KB LDS table; h as bf16 A-fragments in VGPRs, c as
// f32 in D layout; D->A transpose via per-wave LDS slab + s_wait_dscnt.
// ---------------------------------------------------------------------------
__global__ void __launch_bounds__(128, 1)
lstm_kernel(const float* __restrict__ x,      // [N, T]
            const float* __restrict__ W_ih,   // [512]
            const float* __restrict__ b_ih,   // [512]
            const float* __restrict__ b_hh,   // [512]
            const __bf16* __restrict__ pkWhh, // packed B-fragments (global)
            __bf16* __restrict__ hrelu,       // [Npad, 128] bf16
            int N) {
  __shared__ __bf16 sB[WHH_ELEMS];                     // 128 KB weights
  __shared__ __bf16 sH[4][16][HID];                    // 16 KB transpose slabs
  __shared__ float4 sWI[8][16];                        // 2 KB: W_ih per (t8, m)
  __shared__ float4 sBS[8][16];                        // 2 KB: b_ih+b_hh per (t8, m)

  // Stage packed weights into LDS (global_load_b128 -> ds_store_b128).
  {
    const uint4* src = (const uint4*)pkWhh;
    uint4* dst = (uint4*)sB;
#pragma unroll 4
    for (int i = threadIdx.x; i < (WHH_ELEMS * 2) / 16; i += 128) dst[i] = src[i];
  }
  // Stage per-column input weights / fused biases (depend only on t8, m).
  if (threadIdx.x < 128) {
    int t8 = threadIdx.x >> 4, mm = threadIdx.x & 15;
    float wq[4], bq[4];
#pragma unroll
    for (int q = 0; q < 4; ++q) {
      int col = q * HID + t8 * 16 + mm;
      wq[q] = W_ih[col];
      bq[q] = b_ih[col] + b_hh[col];
    }
    sWI[t8][mm] = make_float4(wq[0], wq[1], wq[2], wq[3]);
    sBS[t8][mm] = make_float4(bq[0], bq[1], bq[2], bq[3]);
  }
  __syncthreads();

  const int wave = threadIdx.x >> 5;
  const int lane = threadIdx.x & 31;
  const int m    = lane & 15;
  const int half = lane >> 4;
  const int nodeA = blockIdx.x * MB + wave * 16;       // wave's 16-row base

  // D-layout node ids (row = r + 8*half), clamped for the tail block.
  int ndc[8];
#pragma unroll
  for (int r = 0; r < 8; ++r) {
    int nd = nodeA + r + 8 * half;
    ndc[r] = nd < N ? nd : (N - 1);
  }

  float c[8][8];
#pragma unroll
  for (int t8 = 0; t8 < 8; ++t8)
#pragma unroll
    for (int r = 0; r < 8; ++r) c[t8][r] = 0.0f;

  v16bf a[4];
#pragma unroll
  for (int kc = 0; kc < 4; ++kc)
#pragma unroll
    for (int j = 0; j < 16; ++j) a[kc][j] = f2bf(0.0f);

#pragma unroll 1
  for (int t0 = 0; t0 < TS; t0 += 4) {
    // 4 time steps of x per D-row as one b128 load each.
    float4 xq[8];
#pragma unroll
    for (int r = 0; r < 8; ++r)
      xq[r] = *(const float4*)(x + (long)ndc[r] * TS + t0);

#pragma unroll 1
    for (int tt = 0; tt < 4; ++tt) {
      float xv[8];
#pragma unroll
      for (int r = 0; r < 8; ++r) xv[r] = ((const float*)&xq[r])[tt];

      // Launder loop-variant copies of the LDS indices so LICM cannot hoist
      // the (loop-invariant) LDS loads into hundreds of VGPRs.
      int lofs = lane * 16;
      int ml   = m;
      asm volatile("" : "+v"(lofs), "+v"(ml));
      const __bf16* sBl = sB + lofs;

#pragma unroll
      for (int t8 = 0; t8 < 8; ++t8) {
        float4 wv = sWI[t8][ml];                       // ds_load_b128 (bcast)
        float4 bv = sBS[t8][ml];
        const float* wvp = (const float*)&wv;
        const float* bvp = (const float*)&bv;
        v8f acc[4];
#pragma unroll
        for (int q = 0; q < 4; ++q) {
          const int ntile = q * 8 + t8;
          v16bf bfr[4];
#pragma unroll
          for (int kc = 0; kc < 4; ++kc)
            bfr[kc] = *(const v16bf*)(sBl + (ntile * 4 + kc) * 512);
          // Bias folded into the accumulator init (constant along rows).
          const float bq = bvp[q];
          v8f a0 = {bq, bq, bq, bq, bq, bq, bq, bq};
#pragma unroll
          for (int kc = 0; kc < 4; ++kc)
            a0 = __builtin_amdgcn_wmma_f32_16x16x32_bf16(
                false, a[kc], false, bfr[kc], (short)0, a0, false, false);
          acc[q] = a0;
        }
        // Gate math + state update (D layout: col = t8*16+m, row = r+8*half)
#pragma unroll
        for (int r = 0; r < 8; ++r) {
          float gi = __builtin_fmaf(xv[r], wvp[0], acc[0][r]);
          float gf = __builtin_fmaf(xv[r], wvp[1], acc[1][r]);
          float gg = __builtin_fmaf(xv[r], wvp[2], acc[2][r]);
          float go = __builtin_fmaf(xv[r], wvp[3], acc[3][r]);
          float cn = sig_hw(gf) * c[t8][r] + sig_hw(gi) * tanh_hw(gg);
          float hn = sig_hw(go) * tanh_hw(cn);
          c[t8][r] = cn;
          sH[wave][r + 8 * half][t8 * 16 + m] = (__bf16)hn;  // ds_store_b16
        }
      }

      // D-layout -> A-fragment transpose through the wave-private LDS slab.
      asm volatile("s_wait_dscnt 0" ::: "memory");
      const __bf16* hp = &sH[wave][m][0];
#pragma unroll
      for (int kc = 0; kc < 4; ++kc) {
        v8bf lo = *(const v8bf*)(hp + kc * 32 + 16 * half);     // ds_load_b128
        v8bf hi = *(const v8bf*)(hp + kc * 32 + 16 * half + 8); // ds_load_b128
#pragma unroll
        for (int j = 0; j < 8; ++j) { a[kc][j] = lo[j]; a[kc][8 + j] = hi[j]; }
      }
    }
  }

  // Emit relu(h_T) as bf16 (vectorized b128 moves; lane writes its row half).
  const long rowG = nodeA + m;
  __bf16* orow = hrelu + rowG * HID + half * 64;
  const __bf16* srow = &sH[wave][m][half * 64];
#pragma unroll
  for (int cj = 0; cj < 8; ++cj) {
    v8bf v = *(const v8bf*)(srow + cj * 8);
    v8bf o;
#pragma unroll
    for (int j = 0; j < 8; ++j) {
      float f = bf2f(v[j]);
      o[j] = f2bf(f > 0.0f ? f : 0.0f);
    }
    *(v8bf*)(orow + cj * 8) = o;
  }
}

// ---------------------------------------------------------------------------
// C[Npad,128] = A[Npad,128](bf16) @ B (packed bf16 fragments), f32 out.
// Block = 128 threads (4 waves), each wave one 16-row M tile.
// ---------------------------------------------------------------------------
__global__ void __launch_bounds__(128, 1)
gemm128_kernel(const __bf16* __restrict__ A, const __bf16* __restrict__ pkB,
               float* __restrict__ C) {
  const int wave = threadIdx.x >> 5;
  const int lane = threadIdx.x & 31;
  const int m    = lane & 15;
  const int half = lane >> 4;
  const long rowA = (long)blockIdx.x * MB + wave * 16 + m;
  const __bf16* arow = A + rowA * HID;

  v16bf a[4];
#pragma unroll
  for (int kc = 0; kc < 4; ++kc) {
    v8bf lo = *(const v8bf*)(arow + kc * 32 + 16 * half);
    v8bf hi = *(const v8bf*)(arow + kc * 32 + 16 * half + 8);
#pragma unroll
    for (int j = 0; j < 8; ++j) { a[kc][j] = lo[j]; a[kc][8 + j] = hi[j]; }
  }

  const long rowD = (long)blockIdx.x * MB + wave * 16;
#pragma unroll
  for (int nt = 0; nt < 8; ++nt) {
    v8f acc = {0.f, 0.f, 0.f, 0.f, 0.f, 0.f, 0.f, 0.f};
#pragma unroll
    for (int kc = 0; kc < 4; ++kc) {
      v16bf b = *(const v16bf*)(pkB + ((size_t)(nt * 4 + kc) * 32 + lane) * 16);
      acc = __builtin_amdgcn_wmma_f32_16x16x32_bf16(
          false, a[kc], false, b, (short)0, acc, false, false);
    }
#pragma unroll
    for (int r = 0; r < 8; ++r)
      C[(rowD + r + 8 * half) * HID + nt * 16 + m] = acc[r];
  }
}

// ---------------------------------------------------------------------------
// Graph helpers
// ---------------------------------------------------------------------------
__global__ void fillf_kernel(float* __restrict__ p, float v, long n) {
  long i = (long)blockIdx.x * blockDim.x + threadIdx.x;
  if (i < n) p[i] = v;
}

__global__ void deg_accum_kernel(const int* __restrict__ col,
                                 const float* __restrict__ ew,
                                 float* __restrict__ deg, int E) {
  int e = blockIdx.x * blockDim.x + threadIdx.x;
  if (e < E) atomicAdd(deg + col[e], ew[e]);
}

__global__ void dinv_kernel(const float* __restrict__ deg,
                            float* __restrict__ dinv, int N) {
  int i = blockIdx.x * blockDim.x + threadIdx.x;
  if (i < N) {
    float d = deg[i];
    dinv[i] = d > 0.0f ? rsqrtf(d) : 0.0f;
  }
}

// 32 threads per edge, 4 features each: coalesced reads, clustered atomics.
__global__ void scatter_edges_kernel(const float* __restrict__ xw,
                                     float* __restrict__ out,
                                     const int* __restrict__ row,
                                     const int* __restrict__ col,
                                     const float* __restrict__ ew,
                                     const float* __restrict__ dinv, int E) {
  long id = (long)blockIdx.x * blockDim.x + threadIdx.x;
  int e = (int)(id >> 5);
  if (e >= E) return;
  int f0 = ((int)id & 31) * 4;
  int r = row[e], c = col[e];
  float w = dinv[r] * ew[e] * dinv[c];
  float4 s = *(const float4*)(xw + (long)r * HID + f0);
  float* dst = out + (long)c * HID + f0;
  atomicAdd(dst + 0, w * s.x);
  atomicAdd(dst + 1, w * s.y);
  atomicAdd(dst + 2, w * s.z);
  atomicAdd(dst + 3, w * s.w);
}

__global__ void scatter_self_kernel(const float* __restrict__ xw,
                                    float* __restrict__ out,
                                    const float* __restrict__ dinv, long n) {
  long idx = (long)blockIdx.x * blockDim.x + threadIdx.x;
  if (idx >= n) return;
  int i = (int)(idx >> 7);
  float w = dinv[i] * dinv[i];
  out[idx] += w * xw[idx];
}

__global__ void bias_relu_pack_kernel(const float* __restrict__ in,
                                      const float* __restrict__ bias,
                                      __bf16* __restrict__ outbf, long n) {
  long idx = (long)blockIdx.x * blockDim.x + threadIdx.x;
  if (idx >= n) return;
  int f = (int)(idx & (HID - 1));
  float v = in[idx] + bias[f];
  outbf[idx] = f2bf(v > 0.0f ? v : 0.0f);
}

__global__ void mean_kernel(const float* __restrict__ out2,
                            const float* __restrict__ b2,
                            float* __restrict__ dout, int N) {
  __shared__ float red[256];
  int colf = blockIdx.x;
  int tid = threadIdx.x;
  float s = 0.0f;
  for (int i = tid; i < N; i += 256) s += out2[(long)i * HID + colf];
  red[tid] = s;
  __syncthreads();
  for (int off = 128; off > 0; off >>= 1) {
    if (tid < off) red[tid] += red[tid + off];
    __syncthreads();
  }
  if (tid == 0) dout[colf] = red[0] / (float)N + b2[colf];
}

// ---------------------------------------------------------------------------
// Host launcher
// ---------------------------------------------------------------------------
extern "C" void kernel_launch(void* const* d_in, const int* in_sizes, int n_in,
                              void* d_out, int out_size, void* d_ws, size_t ws_size,
                              hipStream_t stream) {
  (void)n_in; (void)out_size; (void)ws_size;

  const float* node_features = (const float*)d_in[0];   // [N, 128]
  const int*   edge_index    = (const int*)d_in[1];     // [2, E]
  const float* edge_attr     = (const float*)d_in[2];   // [E]
  const float* W_ih          = (const float*)d_in[3];   // [512]
  const float* W_hh          = (const float*)d_in[4];   // [512, 128]
  const float* b_ih          = (const float*)d_in[5];   // [512]
  const float* b_hh          = (const float*)d_in[6];   // [512]
  const float* W1            = (const float*)d_in[7];   // [128, 128]
  const float* b1            = (const float*)d_in[8];   // [128]
  const float* W2            = (const float*)d_in[9];   // [128, 128]
  const float* b2            = (const float*)d_in[10];  // [128]
  float* dout = (float*)d_out;                          // [128]

  const int N = in_sizes[0] / TS;                       // 10000
  const int E = in_sizes[1] / 2;                        // 320000
  const int Npad = ((N + MB - 1) / MB) * MB;            // 10048
  const int row_blocks = Npad / MB;                     // 157
  const int* rowp = edge_index;
  const int* colp = edge_index + E;

  // Workspace partition (256B aligned).
  char* w = (char*)d_ws;
  size_t off = 0;
  auto take = [&](size_t bytes) {
    size_t r = off;
    off = (off + bytes + 255) & ~(size_t)255;
    return r;
  };
  __bf16* pkWhh = (__bf16*)(w + take((size_t)WHH_ELEMS * 2));          // 128 KB
  __bf16* pkW1  = (__bf16*)(w + take((size_t)8 * 4 * 32 * 16 * 2));    // 32 KB
  __bf16* pkW2  = (__bf16*)(w + take((size_t)8 * 4 * 32 * 16 * 2));    // 32 KB
  __bf16* hrelu = (__bf16*)(w + take((size_t)Npad * HID * 2));
  __bf16* x2bf  = (__bf16*)(w + take((size_t)Npad * HID * 2));
  float* xw1  = (float*)(w + take((size_t)Npad * HID * 4));
  float* xw2  = (float*)(w + take((size_t)Npad * HID * 4));
  float* out1 = (float*)(w + take((size_t)Npad * HID * 4));
  float* out2 = (float*)(w + take((size_t)Npad * HID * 4));
  float* deg  = (float*)(w + take((size_t)N * 4));
  float* dinv = (float*)(w + take((size_t)N * 4));

  const long nElem = (long)Npad * HID;
  const int nb = (N + 255) / 256;
  const int fb = (int)((nElem + 255) / 256);
  const long nValid = (long)N * HID;
  const int fbv = (int)((nValid + 255) / 256);
  const int eb1 = (E + 255) / 256;                      // 1 thread / edge
  const int eb32 = (int)(((long)E * 32 + 255) / 256);   // 32 threads / edge

  // --- weight packing ---
  pack_whh_kernel<<<256, 256, 0, stream>>>(W_hh, pkWhh);
  pack_w_kernel<<<64, 256, 0, stream>>>(W1, pkW1);
  pack_w_kernel<<<64, 256, 0, stream>>>(W2, pkW2);

  // --- zero accumulators, self-loop degree init ---
  fillf_kernel<<<fb, 256, 0, stream>>>(out1, 0.0f, nElem);
  fillf_kernel<<<fb, 256, 0, stream>>>(out2, 0.0f, nElem);
  fillf_kernel<<<nb, 256, 0, stream>>>(deg, 1.0f, N);

  // --- LSTM (WMMA, LDS-resident weights, hw tanh) ---
  lstm_kernel<<<row_blocks, 128, 0, stream>>>(node_features, W_ih, b_ih, b_hh,
                                              pkWhh, hrelu, N);

  // --- GCN normalization ---
  deg_accum_kernel<<<eb1, 256, 0, stream>>>(colp, edge_attr, deg, E);
  dinv_kernel<<<nb, 256, 0, stream>>>(deg, dinv, N);

  // --- GCN layer 1 ---
  gemm128_kernel<<<row_blocks, 128, 0, stream>>>(hrelu, pkW1, xw1);
  scatter_edges_kernel<<<eb32, 256, 0, stream>>>(xw1, out1, rowp, colp,
                                                 edge_attr, dinv, E);
  scatter_self_kernel<<<fbv, 256, 0, stream>>>(xw1, out1, dinv, nValid);
  bias_relu_pack_kernel<<<fb, 256, 0, stream>>>(out1, b1, x2bf, nElem);

  // --- GCN layer 2 ---
  gemm128_kernel<<<row_blocks, 128, 0, stream>>>(x2bf, pkW2, xw2);
  scatter_edges_kernel<<<eb32, 256, 0, stream>>>(xw2, out2, rowp, colp,
                                                 edge_attr, dinv, E);
  scatter_self_kernel<<<fbv, 256, 0, stream>>>(xw2, out2, dinv, nValid);

  // --- global mean pool (+ bias b2) ---
  mean_kernel<<<128, 256, 0, stream>>>(out2, b2, dout, N);
}
